// RWKV_x070_37486474559571
// MI455X (gfx1250) — matrix operand order repaired and checked by reading, not compiled, not run on hardware
//
#include <hip/hip_runtime.h>
#include <hip/hip_bf16.h>
#include <math.h>
#include <stdint.h>

// Model dims (fixed by the reference)
#define Tt 1024
#define Cc 2048
#define Hh 32
#define NH 64
#define Vv 32000
#define FFf 8192
#define DWd 64
#define DAd 64
#define DVd 32
#define DGd 128
#define Ll 2

typedef __bf16 bf16_t;
typedef __attribute__((ext_vector_type(16))) __bf16 bf16x16;
typedef __attribute__((ext_vector_type(8)))  __bf16 bf16x8;
typedef __attribute__((ext_vector_type(8)))  float  f32x8;
typedef __attribute__((ext_vector_type(4)))  unsigned int u32x4;
typedef __attribute__((ext_vector_type(8)))  int i32x8;
typedef __attribute__((ext_vector_type(4)))  int i32x4;

#if __has_builtin(__builtin_amdgcn_tensor_load_to_lds)
#define HAVE_TDM 1
#else
#define HAVE_TDM 0
#endif

// ---------------- WMMA fragment loaders (CDNA5 16x16x32 bf16 layouts) -------
// A (16x32, MxK): lane l -> row m=l&15; elems 0..7 : K = (l>>4)*8 + e
//                                       elems 8..15: K = 16 + (l>>4)*8 + (e-8)
__device__ __forceinline__ bf16x16 frag_A(const bf16_t* tile, int lane) {
  int m = lane & 15, half = lane >> 4;
  const bf16_t* rp = tile + m * 32;
  bf16x8 lo = *(const bf16x8*)(rp + half * 8);
  bf16x8 hi = *(const bf16x8*)(rp + 16 + half * 8);
  bf16x16 f;
#pragma unroll
  for (int e = 0; e < 8; ++e) { f[e] = lo[e]; f[e + 8] = hi[e]; }
  return f;
}
// B (32x16, KxN), LDS stored transposed [n][k]: lane l -> col n=l&15;
// elems 0..7 : K = (l>>4)*8 + e; elems 8..15: K = 16 + (l>>4)*8 + (e-8)
__device__ __forceinline__ bf16x16 frag_B(const bf16_t* tile, int lane) {
  int n = lane & 15, k0 = (lane >> 4) * 8;
  const bf16_t* rp = tile + n * 32 + k0;
  bf16x8 lo = *(const bf16x8*)rp;
  bf16x8 hi = *(const bf16x8*)(rp + 16);
  bf16x16 f;
#pragma unroll
  for (int e = 0; e < 8; ++e) { f[e] = lo[e]; f[e + 8] = hi[e]; }
  return f;
}

// TDM: async copy of a 128-row x 32-bf16 tile (row stride K elems) into LDS.
__device__ __forceinline__ void tdm_load_b(const bf16_t* gsrc, unsigned lds,
                                           int N, int K) {
#if HAVE_TDM
  unsigned long long ga = (unsigned long long)(uintptr_t)gsrc;
  u32x4 g0;
  g0[0] = 1u;                                   // count=1 (valid D#)
  g0[1] = lds;                                  // lds_addr (bytes)
  g0[2] = (unsigned)ga;                         // global_addr[31:0]
  g0[3] = ((unsigned)(ga >> 32) & 0x01FFFFFFu)  // global_addr[56:32]
          | (2u << 30);                         // type = 2 ("image")
  i32x8 g1;
  g1[0] = (int)0x00010000u;                     // data_size=1 -> 2 bytes
  g1[1] = (int)((K & 0xFFFFu) << 16);           // tensor_dim0[15:0]
  g1[2] = (int)(((K >> 16) & 0xFFFFu) | ((N & 0xFFFFu) << 16));
  g1[3] = (int)(((N >> 16) & 0xFFFFu) | (32u << 16)); // tile_dim0 = 32
  g1[4] = 128;                                  // tile_dim1 = 128 rows
  g1[5] = K;                                    // tensor_dim0_stride lo32
  g1[6] = 0; g1[7] = 0;
  i32x4 g2 = {0, 0, 0, 0};
  i32x4 g3 = {0, 0, 0, 0};
  i32x8 g4 = {0, 0, 0, 0, 0, 0, 0, 0};
  __builtin_amdgcn_tensor_load_to_lds(g0, g1, g2, g3, g4, 0);
#endif
}

// ---------------- Tiled WMMA GEMM: Out(MxN) (+)= A(MxK) @ Bt(NxK)^T ---------
// 128x128 block tile, 256 threads (8 waves, each 64x32), double-buffered LDS.
// A: f32 activations (converted to bf16 in LDS). Bt: pre-converted bf16
// transposed weights, fetched by the Tensor Data Mover.
// EPI: 0 = store, 1 = accumulate into Out, 2 = relu(x)^2 store
template<int EPI>
__global__ __launch_bounds__(256)
void gemm_wmma(const float* __restrict__ A, const bf16_t* __restrict__ Bt,
               float* __restrict__ Out, int M, int N, int K) {
  __shared__ __align__(16) bf16_t As[2][128 * 32];
  __shared__ __align__(16) bf16_t Bs[2][128 * 32]; // [n][k], k contiguous
  const int tid  = threadIdx.x;
  const int lane = tid & 31;
  const int wave = tid >> 5;
  const int bm = blockIdx.y * 128;
  const int bn = blockIdx.x * 128;
  const int wrow = (wave >> 2) * 64;   // 2 row groups of 64
  const int wcol = (wave & 3) * 32;    // 4 col groups of 32

  f32x8 acc[4][2];
#pragma unroll
  for (int i = 0; i < 4; ++i)
#pragma unroll
    for (int j = 0; j < 2; ++j)
#pragma unroll
      for (int e = 0; e < 8; ++e) acc[i][j][e] = 0.f;

  const int ar = tid >> 1, ac0 = (tid & 1) * 16;     // A tile: 128 rows x 32
  const unsigned ldsB[2] = {(unsigned)(uintptr_t)&Bs[0][0],
                            (unsigned)(uintptr_t)&Bs[1][0]};
  const bf16_t* gB = Bt + (size_t)bn * K;
  const float*  aptr = A + (size_t)(bm + ar) * K + ac0;
  const int niter = K / 32;

  // ---- prologue: fill buffer 0 --------------------------------------------
  if (tid < 32) tdm_load_b(gB, ldsB[0], N, K);
#if !HAVE_TDM
  { int r = tid >> 1, c0 = (tid & 1) * 16;
    const bf16_t* src = Bt + (size_t)(bn + r) * K + c0;
    *(bf16x8*)&Bs[0][r * 32 + c0]     = *(const bf16x8*)src;
    *(bf16x8*)&Bs[0][r * 32 + c0 + 8] = *(const bf16x8*)(src + 8); }
#endif
  {
    float4 f0 = *(const float4*)(aptr + 0);
    float4 f1 = *(const float4*)(aptr + 4);
    float4 f2 = *(const float4*)(aptr + 8);
    float4 f3 = *(const float4*)(aptr + 12);
    bf16_t* d = &As[0][ar * 32 + ac0];
    d[0] = (bf16_t)f0.x; d[1]  = (bf16_t)f0.y; d[2]  = (bf16_t)f0.z; d[3]  = (bf16_t)f0.w;
    d[4] = (bf16_t)f1.x; d[5]  = (bf16_t)f1.y; d[6]  = (bf16_t)f1.z; d[7]  = (bf16_t)f1.w;
    d[8] = (bf16_t)f2.x; d[9]  = (bf16_t)f2.y; d[10] = (bf16_t)f2.z; d[11] = (bf16_t)f2.w;
    d[12]= (bf16_t)f3.x; d[13] = (bf16_t)f3.y; d[14] = (bf16_t)f3.z; d[15] = (bf16_t)f3.w;
  }
#if HAVE_TDM
  if (tid < 32) __builtin_amdgcn_s_wait_tensorcnt(0);
#endif
  __syncthreads();

  // ---- main loop: compute buf, prefetch buf^1 -----------------------------
  for (int i = 0; i < niter; ++i) {
    const int buf = i & 1;
    const bool pre = (i + 1 < niter);
    float4 f0, f1, f2, f3;
    if (pre) {
      if (tid < 32) tdm_load_b(gB + (i + 1) * 32, ldsB[buf ^ 1], N, K);
#if !HAVE_TDM
      { int r = tid >> 1, c0 = (tid & 1) * 16;
        const bf16_t* src = Bt + (size_t)(bn + r) * K + (i + 1) * 32 + c0;
        *(bf16x8*)&Bs[buf ^ 1][r * 32 + c0]     = *(const bf16x8*)src;
        *(bf16x8*)&Bs[buf ^ 1][r * 32 + c0 + 8] = *(const bf16x8*)(src + 8); }
#endif
      const float* ap = aptr + (size_t)(i + 1) * 32;
      __builtin_prefetch(ap + 32, 0, 3);
      f0 = *(const float4*)(ap + 0);
      f1 = *(const float4*)(ap + 4);
      f2 = *(const float4*)(ap + 8);
      f3 = *(const float4*)(ap + 12);
    }
    // compute on buffer `buf`
    bf16x16 af[4], bfv[2];
#pragma unroll
    for (int ti = 0; ti < 4; ++ti)
      af[ti] = frag_A(&As[buf][(wrow + ti * 16) * 32], lane);
#pragma unroll
    for (int tj = 0; tj < 2; ++tj)
      bfv[tj] = frag_B(&Bs[buf][(wcol + tj * 16) * 32], lane);
#pragma unroll
    for (int ti = 0; ti < 4; ++ti)
#pragma unroll
      for (int tj = 0; tj < 2; ++tj)
        acc[ti][tj] = __builtin_amdgcn_wmma_f32_16x16x32_bf16(
            false, af[ti], false, bfv[tj], (short)0, acc[ti][tj], false, false);
    if (pre) {
      bf16_t* d = &As[buf ^ 1][ar * 32 + ac0];
      d[0] = (bf16_t)f0.x; d[1]  = (bf16_t)f0.y; d[2]  = (bf16_t)f0.z; d[3]  = (bf16_t)f0.w;
      d[4] = (bf16_t)f1.x; d[5]  = (bf16_t)f1.y; d[6]  = (bf16_t)f1.z; d[7]  = (bf16_t)f1.w;
      d[8] = (bf16_t)f2.x; d[9]  = (bf16_t)f2.y; d[10] = (bf16_t)f2.z; d[11] = (bf16_t)f2.w;
      d[12]= (bf16_t)f3.x; d[13] = (bf16_t)f3.y; d[14] = (bf16_t)f3.z; d[15] = (bf16_t)f3.w;
    }
#if HAVE_TDM
    if (tid < 32) __builtin_amdgcn_s_wait_tensorcnt(0);
#endif
    __syncthreads();  // publish buf^1; retire all reads of buf
  }

  // epilogue: C/D layout row = vgpr + 8*(l>>4), col = l&15; immediate offsets
  const int mo = (lane >> 4) * 8;
  const int no = lane & 15;
  float* ob = Out + (size_t)(bm + wrow + mo) * N + bn + wcol + no;
#pragma unroll
  for (int ti = 0; ti < 4; ++ti)
#pragma unroll
    for (int tj = 0; tj < 2; ++tj)
#pragma unroll
      for (int j = 0; j < 8; ++j) {
        size_t o = (size_t)(ti * 16 + j) * N + tj * 16;
        float v = acc[ti][tj][j];
        if (EPI == 0)      ob[o] = v;
        else if (EPI == 1) ob[o] += v;
        else { float r = fmaxf(v, 0.f); ob[o] = r * r; }
      }
}

// ---------------- one-time weight convert+transpose: W(KxN) -> Wt(NxK) bf16 -
__global__ __launch_bounds__(256)
void transpose_bf16_k(const float* __restrict__ W, bf16_t* __restrict__ Wt,
                      int Kd, int Nd) {
  __shared__ float tile[32][33];
  int bk = blockIdx.x * 32;
  int bn = blockIdx.y * 32;
  int tx = threadIdx.x & 31, ty = threadIdx.x >> 5; // 32x8
#pragma unroll
  for (int r = 0; r < 32; r += 8)
    tile[ty + r][tx] = W[(size_t)(bk + ty + r) * Nd + bn + tx];
  __syncthreads();
#pragma unroll
  for (int r = 0; r < 32; r += 8)
    Wt[(size_t)(bn + ty + r) * Kd + bk + tx] = (bf16_t)tile[tx][ty + r];
}

// ---------------- Elementwise / small kernels --------------------------------
__global__ __launch_bounds__(256)
void embed_k(const int* __restrict__ idx, const float* __restrict__ emb,
             float* __restrict__ X) {
  size_t i = (size_t)blockIdx.x * 256 + threadIdx.x;
  int t = (int)(i / Cc), c = (int)(i % Cc);
  X[i] = emb[(size_t)idx[t] * Cc + c];
}

__global__ __launch_bounds__(256)
void layernorm_k(const float* __restrict__ x, const float* __restrict__ w,
                 const float* __restrict__ b, float* __restrict__ out, float eps) {
  __shared__ float red[256];
  int t = blockIdx.x, tid = threadIdx.x;
  const float* row = x + (size_t)t * Cc;
  float s = 0.f;
  for (int c = tid; c < Cc; c += 256) s += row[c];
  red[tid] = s; __syncthreads();
  for (int k = 128; k > 0; k >>= 1) { if (tid < k) red[tid] += red[tid + k]; __syncthreads(); }
  float mu = red[0] * (1.f / Cc); __syncthreads();
  float vs = 0.f;
  for (int c = tid; c < Cc; c += 256) { float d = row[c] - mu; vs += d * d; }
  red[tid] = vs; __syncthreads();
  for (int k = 128; k > 0; k >>= 1) { if (tid < k) red[tid] += red[tid + k]; __syncthreads(); }
  float inv = rsqrtf(red[0] * (1.f / Cc) + eps);
  float* orow = out + (size_t)t * Cc;
  for (int c = tid; c < Cc; c += 256) orow[c] = (row[c] - mu) * inv * w[c] + b[c];
}

__global__ __launch_bounds__(256)
void timemix_k(const float* __restrict__ XX,
               const float* __restrict__ mr, const float* __restrict__ mw,
               const float* __restrict__ mk, const float* __restrict__ mv,
               const float* __restrict__ ma, const float* __restrict__ mg,
               float* XR, float* XW, float* XK, float* XV, float* XA, float* XG) {
  size_t i = (size_t)blockIdx.x * 256 + threadIdx.x;
  int t = (int)(i / Cc), c = (int)(i % Cc);
  float x = XX[i];
  float prev = t ? XX[i - Cc] : 0.f;
  float sh = prev - x;
  XR[i] = x + sh * mr[c]; XW[i] = x + sh * mw[c]; XK[i] = x + sh * mk[c];
  XV[i] = x + sh * mv[c]; XA[i] = x + sh * ma[c]; XG[i] = x + sh * mg[c];
}

__global__ __launch_bounds__(256)
void cmix_k(const float* __restrict__ XX, const float* __restrict__ mk,
            float* __restrict__ OutB) {
  size_t i = (size_t)blockIdx.x * 256 + threadIdx.x;
  int t = (int)(i / Cc), c = (int)(i % Cc);
  float x = XX[i];
  float prev = t ? XX[i - Cc] : 0.f;
  OutB[i] = x + (prev - x) * mk[c];
}

// out = act(bias + X @ W); act: 0 none, 1 tanh, 2 sigmoid
__global__ __launch_bounds__(256)
void small_gemm_k(const float* __restrict__ X, const float* __restrict__ W,
                  const float* __restrict__ bias, float* __restrict__ out,
                  int Kd, int Nd, int act) {
  int i = blockIdx.x * 256 + threadIdx.x;
  if (i >= Tt * Nd) return;
  int t = i / Nd, n = i % Nd;
  const float* xr = X + (size_t)t * Kd;
  float s = 0.f;
  for (int k = 0; k < Kd; ++k) s += xr[k] * W[(size_t)k * Nd + n];
  if (bias) s += bias[n];
  if (act == 1) s = tanhf(s);
  else if (act == 2) s = 1.f / (1.f + expf(-s));
  out[i] = s;
}

__device__ __forceinline__ float reduce64(float* red, int n, float v) {
  red[n] = v; __syncthreads();
#pragma unroll
  for (int s = 32; s > 0; s >>= 1) { if (n < s) red[n] += red[n + s]; __syncthreads(); }
  float r = red[0]; __syncthreads();
  return r;
}

// per-(t,h): kk-normalize, k/v rewrite, decay, aa/bb
__global__ __launch_bounds__(64)
void prep_wkv_k(float* __restrict__ Kb, float* __restrict__ Vb, float* __restrict__ Wd,
                const float* __restrict__ Ab, const float* __restrict__ kkw,
                const float* __restrict__ kaw, const float* __restrict__ vmix,
                float* __restrict__ VF, float* __restrict__ AA, float* __restrict__ BB,
                int is_layer0) {
  __shared__ float red[64];
  int t = blockIdx.x, h = blockIdx.y, n = threadIdx.x;
  int c = h * NH + n;
  size_t idx = (size_t)t * Cc + c;
  float kraw = Kb[idx];
  float kk = kraw * kkw[c];
  float ss = reduce64(red, n, kk * kk);
  float kkn = kk / fmaxf(sqrtf(ss), 1e-12f);
  float a = Ab[idx];
  Kb[idx] = kraw * (1.f + (a - 1.f) * kaw[c]);
  float v = Vb[idx];
  if (is_layer0) { VF[idx] = v; }
  else { v = v + (VF[idx] - v) * vmix[idx]; Vb[idx] = v; }
  float wv = Wd[idx];                       // w0 + wl
  float wf = -log1pf(expf(-wv)) - 0.5f;     // -softplus(-x) - 0.5
  Wd[idx] = expf(-expf(wf));                // decay
  AA[idx] = -kkn;
  BB[idx] = kkn * a;
}

// sequential recurrence: one block per head, thread i owns state row i (64 regs)
__global__ __launch_bounds__(64)
void wkv7_k(const float* __restrict__ Rb, const float* __restrict__ Wd,
            const float* __restrict__ Kb, const float* __restrict__ Vb,
            const float* __restrict__ AA, const float* __restrict__ BB,
            float* __restrict__ Y) {
  int h = blockIdx.x, i = threadIdx.x;
  __shared__ float rs[64], ws[64], ks[64], vs[64], as[64], bs[64];
  float S[64];
#pragma unroll
  for (int j = 0; j < 64; ++j) S[j] = 0.f;
  for (int t = 0; t < Tt; ++t) {
    size_t base = (size_t)t * Cc + h * NH;
    rs[i] = Rb[base + i]; ws[i] = Wd[base + i]; ks[i] = Kb[base + i];
    vs[i] = Vb[base + i]; as[i] = AA[base + i]; bs[i] = BB[base + i];
    __syncthreads();
    float sa = 0.f;
#pragma unroll
    for (int j = 0; j < 64; ++j) sa += S[j] * as[j];
    float vi = vs[i];
    float y = 0.f;
#pragma unroll
    for (int j = 0; j < 64; ++j) {
      float s = S[j] * ws[j] + sa * bs[j] + vi * ks[j];
      S[j] = s;
      y += s * rs[j];
    }
    Y[base + i] = y;
    __syncthreads();
  }
}

// group-norm + (r*k*r_k).sum * v bonus + gate by g
__global__ __launch_bounds__(64)
void gn_bonus_g_k(const float* __restrict__ Y, const float* __restrict__ gw,
                  const float* __restrict__ gb, const float* __restrict__ R,
                  const float* __restrict__ Kv, const float* __restrict__ Vx,
                  const float* __restrict__ rk, const float* __restrict__ G,
                  float* __restrict__ YG) {
  __shared__ float red[64];
  int t = blockIdx.x, h = blockIdx.y, n = threadIdx.x;
  int c = h * NH + n;
  size_t idx = (size_t)t * Cc + c;
  float y = Y[idx];
  float mu = reduce64(red, n, y) * (1.f / NH);
  float d = y - mu;
  float var = reduce64(red, n, d * d) * (1.f / NH);
  float yn = d * rsqrtf(var + 6.4e-4f) * gw[c] + gb[c];
  float s = reduce64(red, n, R[idx] * Kv[idx] * rk[c]);
  YG[idx] = (yn + s * Vx[idx]) * G[idx];
}

// ---------------- Host orchestration ----------------------------------------
extern "C" void kernel_launch(void* const* d_in, const int* in_sizes, int n_in,
                              void* d_out, int out_size, void* d_ws, size_t ws_size,
                              hipStream_t stream) {
  const int*   idx   = (const int*)  d_in[0];
  const float* emb   = (const float*)d_in[1];
  const float* ln1w  = (const float*)d_in[2];
  const float* ln1b  = (const float*)d_in[3];
  const float* ln2w  = (const float*)d_in[4];
  const float* ln2b  = (const float*)d_in[5];
  const float* mxr   = (const float*)d_in[6];
  const float* mxw   = (const float*)d_in[7];
  const float* mxk   = (const float*)d_in[8];
  const float* mxv   = (const float*)d_in[9];
  const float* mxa   = (const float*)d_in[10];
  const float* mxg   = (const float*)d_in[11];
  const float* w0    = (const float*)d_in[12];
  const float* w1    = (const float*)d_in[13];
  const float* w2    = (const float*)d_in[14];
  const float* a0    = (const float*)d_in[15];
  const float* a1    = (const float*)d_in[16];
  const float* a2    = (const float*)d_in[17];
  const float* v0    = (const float*)d_in[18];
  const float* v1    = (const float*)d_in[19];
  const float* v2    = (const float*)d_in[20];
  const float* g1    = (const float*)d_in[21];
  const float* g2    = (const float*)d_in[22];
  const float* k_k   = (const float*)d_in[23];
  const float* k_a   = (const float*)d_in[24];
  const float* r_k   = (const float*)d_in[25];
  const float* Rw    = (const float*)d_in[26];
  const float* Kw    = (const float*)d_in[27];
  const float* Vw    = (const float*)d_in[28];
  const float* Ow    = (const float*)d_in[29];
  const float* lnxw  = (const float*)d_in[30];
  const float* lnxb  = (const float*)d_in[31];
  const float* fxk   = (const float*)d_in[32];
  const float* fK    = (const float*)d_in[33];
  const float* fV    = (const float*)d_in[34];
  const float* lnow  = (const float*)d_in[35];
  const float* lnob  = (const float*)d_in[36];
  const float* headw = (const float*)d_in[37];

  const size_t TC = (size_t)Tt * Cc;
  const size_t CC = (size_t)Cc * Cc;
  const size_t CF = (size_t)Cc * FFf;
  float* ws = (float*)d_ws;
  float* X    = ws + 0 * TC;
  float* XX   = ws + 1 * TC;
  float* XR   = ws + 2 * TC;
  float* XW   = ws + 3 * TC;
  float* XK   = ws + 4 * TC;
  float* XV   = ws + 5 * TC;
  float* XA   = ws + 6 * TC;
  float* XG   = ws + 7 * TC;
  float* Rb   = ws + 8 * TC;
  float* Kb   = ws + 9 * TC;
  float* Vb   = ws + 10 * TC;
  float* Wd   = ws + 11 * TC;
  float* Ab   = ws + 12 * TC;
  float* Gb   = ws + 13 * TC;
  float* VF   = ws + 14 * TC;
  float* AA   = ws + 15 * TC;
  float* BB   = ws + 16 * TC;
  float* Yb   = ws + 17 * TC;
  float* TMP1 = ws + 18 * TC;         // T x <=128
  float* TMP2 = ws + 19 * TC;         // T x C (vmix / yg / kf_in)
  float* KF   = ws + 20 * TC;         // T x FF = 4*TC
  // bf16 transposed-weight cache (after 24*TC floats)
  bf16_t* wb   = (bf16_t*)(ws + 24 * TC);
  bf16_t* RwT  = wb;                  // [L][C][C]
  bf16_t* KwT  = RwT + Ll * CC;
  bf16_t* VwT  = KwT + Ll * CC;
  bf16_t* OwT  = VwT + Ll * CC;
  bf16_t* fKT  = OwT + Ll * CC;       // [L][FF][C]
  bf16_t* fVT  = fKT + Ll * CF;       // [L][C][FF]
  bf16_t* hT   = fVT + Ll * CF;       // [V][C]

  const dim3 blk256(256), blk64(64);
  const dim3 grdTC((unsigned)(TC / 256));
  const dim3 grdTH(Tt, Hh);
  const dim3 gCC(Cc / 128, Tt / 128);   // GEMM T x C
  const dim3 gFF(FFf / 128, Tt / 128);  // GEMM T x FF
  const dim3 gV(Vv / 128, Tt / 128);    // GEMM T x V

  // ---- one-time (per call) weight convert + transpose to bf16 -------------
  for (int i = 0; i < Ll; ++i) {
    dim3 gT(Cc / 32, Cc / 32);
    transpose_bf16_k<<<gT, blk256, 0, stream>>>(Rw + i * CC, RwT + i * CC, Cc, Cc);
    transpose_bf16_k<<<gT, blk256, 0, stream>>>(Kw + i * CC, KwT + i * CC, Cc, Cc);
    transpose_bf16_k<<<gT, blk256, 0, stream>>>(Vw + i * CC, VwT + i * CC, Cc, Cc);
    transpose_bf16_k<<<gT, blk256, 0, stream>>>(Ow + i * CC, OwT + i * CC, Cc, Cc);
    dim3 gTf(Cc / 32, FFf / 32), gTv(FFf / 32, Cc / 32);
    transpose_bf16_k<<<gTf, blk256, 0, stream>>>(fK + i * CF, fKT + i * CF, Cc, FFf);
    transpose_bf16_k<<<gTv, blk256, 0, stream>>>(fV + i * CF, fVT + i * CF, FFf, Cc);
  }
  transpose_bf16_k<<<dim3(Cc / 32, Vv / 32), blk256, 0, stream>>>(headw, hT, Cc, Vv);

  embed_k<<<grdTC, blk256, 0, stream>>>(idx, emb, X);

  for (int i = 0; i < Ll; ++i) {
    // ---- time mix ----
    layernorm_k<<<Tt, blk256, 0, stream>>>(X, ln1w + i * Cc, ln1b + i * Cc, XX, 1e-5f);
    timemix_k<<<grdTC, blk256, 0, stream>>>(XX, mxr + i * Cc, mxw + i * Cc, mxk + i * Cc,
                                            mxv + i * Cc, mxa + i * Cc, mxg + i * Cc,
                                            XR, XW, XK, XV, XA, XG);
    gemm_wmma<0><<<gCC, blk256, 0, stream>>>(XR, RwT + i * CC, Rb, Tt, Cc, Cc);
    gemm_wmma<0><<<gCC, blk256, 0, stream>>>(XK, KwT + i * CC, Kb, Tt, Cc, Cc);
    gemm_wmma<0><<<gCC, blk256, 0, stream>>>(XV, VwT + i * CC, Vb, Tt, Cc, Cc);
    // wl = tanh(xw@w1) @ w2 + w0  (into Wd)
    small_gemm_k<<<(Tt * DWd + 255) / 256, blk256, 0, stream>>>(
        XW, w1 + (size_t)i * Cc * DWd, nullptr, TMP1, Cc, DWd, 1);
    small_gemm_k<<<(Tt * Cc + 255) / 256, blk256, 0, stream>>>(
        TMP1, w2 + (size_t)i * DWd * Cc, w0 + i * Cc, Wd, DWd, Cc, 0);
    // a = sigmoid(a0 + (xa@a1)@a2)
    small_gemm_k<<<(Tt * DAd + 255) / 256, blk256, 0, stream>>>(
        XA, a1 + (size_t)i * Cc * DAd, nullptr, TMP1, Cc, DAd, 0);
    small_gemm_k<<<(Tt * Cc + 255) / 256, blk256, 0, stream>>>(
        TMP1, a2 + (size_t)i * DAd * Cc, a0 + i * Cc, Ab, DAd, Cc, 2);
    // g = sigmoid(xg@g1) @ g2
    small_gemm_k<<<(Tt * DGd + 255) / 256, blk256, 0, stream>>>(
        XG, g1 + (size_t)i * Cc * DGd, nullptr, TMP1, Cc, DGd, 2);
    small_gemm_k<<<(Tt * Cc + 255) / 256, blk256, 0, stream>>>(
        TMP1, g2 + (size_t)i * DGd * Cc, nullptr, Gb, DGd, Cc, 0);
    if (i > 0) {  // v-mix gate = sigmoid(v0 + (xv@v1)@v2)
      small_gemm_k<<<(Tt * DVd + 255) / 256, blk256, 0, stream>>>(
          XV, v1 + (size_t)i * Cc * DVd, nullptr, TMP1, Cc, DVd, 0);
      small_gemm_k<<<(Tt * Cc + 255) / 256, blk256, 0, stream>>>(
          TMP1, v2 + (size_t)i * DVd * Cc, v0 + i * Cc, TMP2, DVd, Cc, 2);
    }
    prep_wkv_k<<<grdTH, blk64, 0, stream>>>(Kb, Vb, Wd, Ab, k_k + i * Cc, k_a + i * Cc,
                                            TMP2, VF, AA, BB, i == 0 ? 1 : 0);
    wkv7_k<<<Hh, blk64, 0, stream>>>(Rb, Wd, Kb, Vb, AA, BB, Yb);
    gn_bonus_g_k<<<grdTH, blk64, 0, stream>>>(Yb, lnxw + i * Cc, lnxb + i * Cc,
                                              Rb, Kb, Vb, r_k + i * Cc, Gb, TMP2);
    gemm_wmma<1><<<gCC, blk256, 0, stream>>>(TMP2, OwT + i * CC, X, Tt, Cc, Cc);
    // ---- channel mix ----
    layernorm_k<<<Tt, blk256, 0, stream>>>(X, ln2w + i * Cc, ln2b + i * Cc, XX, 1e-5f);
    cmix_k<<<grdTC, blk256, 0, stream>>>(XX, fxk + i * Cc, TMP2);
    gemm_wmma<2><<<gFF, blk256, 0, stream>>>(TMP2, fKT + i * CF, KF, Tt, FFf, Cc);
    gemm_wmma<1><<<gCC, blk256, 0, stream>>>(KF, fVT + i * CF, X, Tt, Cc, FFf);
  }

  layernorm_k<<<Tt, blk256, 0, stream>>>(X, lnow, lnob, XX, 1e-5f);
  gemm_wmma<0><<<gV, blk256, 0, stream>>>(XX, hT, (float*)d_out, Tt, Vv, Cc);

  (void)in_sizes; (void)n_in; (void)out_size; (void)ws_size;
}
